// SequencePredictor_25821343383657
// MI455X (gfx1250) — matrix operand order repaired and checked by reading, not compiled, not run on hardware
//
#include <hip/hip_runtime.h>
#include <hip/hip_bf16.h>

// ---------------------------------------------------------------------------
// Problem dims
//  x: (128,256,8,32) f32 -> bilinear x2 -> conv3x3(256->256)+ReLU+pool2 ->
//  enc (128,256 pos, 296 ch) -> 32-step Bahdanau attention + GRU + NLL loss.
//
// Design (MI455X / gfx1250, wave32, WMMA f32_16x16x32_f16):
//  * conv+upsample folded into 4 parity 3x3 convs on the coarse 8x32 grid;
//    pool = max over parities in the GEMM epilogue. M=32768,N=256,K=2304.
//  * encoder attention projection hoisted out of the time scan.
//  * all dense math via v_wmma_f32_16x16x32_f16, f32 accumulation.
// ---------------------------------------------------------------------------

typedef _Float16 h16;
typedef __attribute__((ext_vector_type(16))) _Float16 v16h;
typedef __attribute__((ext_vector_type(8)))  float    v8f;

#define B_      128
#define HID_    256
#define NC_     163
#define T_      32
#define L_      256      // 8*32 attention positions
#define CE_     296      // encoder channels
#define CEP_    320      // padded to 10 k-tiles
#define KCONV_  2304     // 256*9
#define RINP_   576      // 552 padded to 18 k-tiles
#define G3_     768      // 3*HID
#define NCP_    176      // 163 padded to 11 n-tiles

// ---------------- WMMA fragment loaders (wave32, contiguous K=16g..16g+15 per lane half) -----
__device__ __forceinline__ v16h frag_a_rowmajor(const h16* Abase, int ldk, int lane) {
  // A 16x32 f16: lane -> M=lane&15; halves = K=16g..16g+15
  int row = lane & 15, grp = lane >> 4;
  const h16* p0 = Abase + (size_t)row * ldk + 16 * grp;
  v16h a;
#pragma unroll
  for (int j = 0; j < 16; ++j) a[j] = p0[j];
  return a;
}

__device__ __forceinline__ v16h frag_b_nmajor(const h16* Bt, int ldk, int lane) {
  // B 32x16 f16, Bt is N-major [n][k]: lane -> N=lane&15, halves = K=16g..16g+15
  int col = lane & 15, grp = lane >> 4;
  const h16* p = Bt + (size_t)col * ldk + 16 * grp;
  v16h b;
#pragma unroll
  for (int j = 0; j < 16; ++j) b[j] = p[j];
  return b;
}

// ---------------- generic 16x16-tile GEMM: one wave per tile ----------------
// C[m0..m0+15, n0..n0+15] = A(row-major,[M][ldk]) * Bt(N-major,[N][ldk]) (+bias)
__global__ void gemm16_k(const h16* __restrict__ A, const h16* __restrict__ Bt,
                         float* __restrict__ C, const float* __restrict__ bias,
                         int ldk, int ldc, int ktiles) {
  int lane = threadIdx.x & 31;
  int m0 = blockIdx.x * 16, n0 = blockIdx.y * 16;
  const h16* Arow = A + (size_t)m0 * ldk;
  const h16* Bcol = Bt + (size_t)n0 * ldk;
  v8f acc = {};
  for (int kt = 0; kt < ktiles; ++kt) {
    v16h a = frag_a_rowmajor(Arow + kt * 32, ldk, lane);
    v16h b = frag_b_nmajor(Bcol + kt * 32, ldk, lane);
    acc = __builtin_amdgcn_wmma_f32_16x16x32_f16(false, a, false, b, (short)0,
                                                 acc, false, false);
  }
  int n = n0 + (lane & 15);
  int mrow = m0 + ((lane >> 4) << 3);
  float bv = bias ? bias[n] : 0.f;
#pragma unroll
  for (int j = 0; j < 8; ++j)
    C[(size_t)(mrow + j) * ldc + n] = acc[j] + bv;
}

// ---------------- conv (folded bilinear) implicit GEMM ----------------------
// Block: 256 thr = 8 waves. blockIdx.x = M-tile (16 pooled pixels, same b,py),
// blockIdx.y in {0,1}: wave w handles n0=(y*8+w)*16. 4 parity accumulators,
// epilogue: relu(max_p + bias) -> enc f16 [b][l][c], stride CEP_.
__global__ void conv_gemm_k(const h16* __restrict__ xh, const h16* __restrict__ wparT,
                            const float* __restrict__ conv_b, h16* __restrict__ enc) {
  __shared__ h16 lA[16 * 32];
  int tid = threadIdx.x, lane = tid & 31, wave = tid >> 5;
  int m0 = blockIdx.x * 16;
  int b   = m0 >> 8;
  int py  = (m0 & 255) >> 5;
  int px0 = m0 & 31;                 // 0 or 16
  int n0  = (blockIdx.y * 8 + wave) * 16;
  v8f acc[4] = {};
  for (int kb = 0; kb < KCONV_; kb += 32) {
    // stage im2col A tile (16 rows x 32 k) in LDS
    for (int e = tid; e < 512; e += 256) {
      int row = e >> 5, kk = e & 31, k = kb + kk;
      int ic = k / 9, r = k - ic * 9;
      int cy = r / 3, cx = r - cy * 3;
      int iy = py + cy - 1, ix = px0 + row + cx - 1;
      h16 v = (h16)0.f;
      if (iy >= 0 && iy < 8 && ix >= 0 && ix < 32)
        v = xh[(((size_t)b * 256 + ic) * 8 + iy) * 32 + ix];
      lA[row * 32 + kk] = v;
    }
    __syncthreads();
    v16h a = frag_a_rowmajor(lA, 32, lane);
#pragma unroll
    for (int p = 0; p < 4; ++p) {
      const h16* Bc = wparT + ((size_t)(p * 256 + n0)) * KCONV_ + kb;
      v16h bf = frag_b_nmajor(Bc, KCONV_, lane);
      acc[p] = __builtin_amdgcn_wmma_f32_16x16x32_f16(false, a, false, bf,
                                                      (short)0, acc[p], false, false);
    }
    __syncthreads();
  }
  int n = n0 + (lane & 15), grp = lane >> 4;
  float bias = conv_b[n];
#pragma unroll
  for (int j = 0; j < 8; ++j) {
    int row = j + 8 * grp;
    float v = fmaxf(fmaxf(acc[0][j], acc[1][j]), fmaxf(acc[2][j], acc[3][j])) + bias;
    v = fmaxf(v, 0.f);
    int l = py * 32 + px0 + row;
    enc[((size_t)b * L_ + l) * CEP_ + n] = (h16)v;
  }
}

// ---------------- prep kernels ---------------------------------------------
__global__ void cast_x_k(const float* __restrict__ x, h16* __restrict__ xh, int n) {
  int i = blockIdx.x * 256 + threadIdx.x;
  if (i < n) xh[i] = (h16)x[i];
}

// bilinear(2x, half-pixel) folded into per-parity 3x3 weights
__device__ const float BLW[2][3][3] = {
  {{0.75f, 0.25f, 0.f}, {0.25f, 0.75f, 0.f}, {0.f, 0.75f, 0.25f}},   // parity 0
  {{0.25f, 0.75f, 0.f}, {0.f, 0.75f, 0.25f}, {0.f, 0.25f, 0.75f}}};  // parity 1

__global__ void wpar_k(const float* __restrict__ conv_w, h16* __restrict__ wparT) {
  int idx = blockIdx.x * 256 + threadIdx.x;          // 4*256*256*9
  if (idx >= 4 * 256 * 256 * 9) return;
  int k9 = idx % 9;  int t = idx / 9;
  int ic = t % 256;  t /= 256;
  int oc = t % 256;  int p = t / 256;                // 0..3
  int cy = k9 / 3, cx = k9 - cy * 3;
  int ppy = p >> 1, ppx = p & 1;
  float s = 0.f;
#pragma unroll
  for (int fy = 0; fy < 3; ++fy)
#pragma unroll
    for (int fx = 0; fx < 3; ++fx)
      s += conv_w[((oc * 256 + ic) * 3 + fy) * 3 + fx] * BLW[ppy][fy][cy] * BLW[ppx][fx][cx];
  wparT[((size_t)(p * 256 + oc)) * KCONV_ + ic * 9 + k9] = (h16)s;
}

// generic f32 -> f16 N-major repack with zero padding
__global__ void pack_f16_k(const float* __restrict__ src, h16* __restrict__ dst,
                           int N, int Nsrc, int Kp, int Ksrc, int srcld, int koff) {
  int idx = blockIdx.x * 256 + threadIdx.x;
  if (idx >= N * Kp) return;
  int n = idx / Kp, k = idx - n * Kp;
  float v = (n < Nsrc && k < Ksrc) ? src[(size_t)n * srcld + koff + k] : 0.f;
  dst[idx] = (h16)v;
}

// wordemb[tok][h] = emb_dec[tok] . word_w[h] + word_b[h]   (163 x 256, tiny)
__global__ void wordemb_k(const float* __restrict__ emb_dec, const float* __restrict__ word_w,
                          const float* __restrict__ word_b, float* __restrict__ we) {
  int tok = blockIdx.x, h = threadIdx.x;
  float s = word_b[h];
  for (int c = 0; c < NC_; ++c) s += emb_dec[tok * NC_ + c] * word_w[h * NC_ + c];
  we[tok * HID_ + h] = s;
}

// positional channels 256..295 and zero pad 296..319 of enc
__global__ void posfill_k(const float* __restrict__ x_emb, const float* __restrict__ y_emb,
                          h16* __restrict__ enc) {
  int idx = blockIdx.x * 256 + threadIdx.x;          // 128*256*64
  int c64 = idx % 64; int t = idx / 64;
  int l = t % L_; int b = t / L_;
  if (b >= B_) return;
  float v = 0.f;
  if (c64 < 32)       v = x_emb[(l & 31) * 32 + c64];
  else if (c64 < 40)  v = y_emb[(l >> 5) * 8 + (c64 - 32)];
  enc[((size_t)b * L_ + l) * CEP_ + 256 + c64] = (h16)v;
}

// ---------------- per-step kernels -----------------------------------------
__global__ void scores_k(const float* __restrict__ encA, const float* __restrict__ hterm,
                         const float* __restrict__ attn_v, float* __restrict__ scores) {
  int b = blockIdx.x, lane = threadIdx.x & 31, wave = threadIdx.x >> 5;
  const float* ht = hterm + b * HID_;
  for (int l = wave; l < L_; l += 8) {
    const float* ea = encA + ((size_t)b * L_ + l) * HID_;
    float s = 0.f;
    for (int h = lane; h < HID_; h += 32)
      s += tanhf(ea[h] + ht[h]) * attn_v[h];
#pragma unroll
    for (int off = 16; off; off >>= 1) s += __shfl_xor(s, off, 32);
    if (lane == 0) scores[b * L_ + l] = s;
  }
}

__global__ void softctx_k(const float* __restrict__ scores, const h16* __restrict__ enc,
                          const float* __restrict__ wordemb, const int* __restrict__ dec_tgt,
                          int t, h16* __restrict__ rin) {
  __shared__ float w[L_];
  __shared__ float red[256];
  int b = blockIdx.x, tid = threadIdx.x;
  float s = scores[b * L_ + tid];
  red[tid] = s; __syncthreads();
  for (int off = 128; off; off >>= 1) { if (tid < off) red[tid] = fmaxf(red[tid], red[tid + off]); __syncthreads(); }
  float mx = red[0]; __syncthreads();
  float e = expf(s - mx);
  red[tid] = e; __syncthreads();
  for (int off = 128; off; off >>= 1) { if (tid < off) red[tid] += red[tid + off]; __syncthreads(); }
  float inv = 1.f / red[0];
  w[tid] = e * inv;
  __syncthreads();
  int tok = (t == 0) ? 0 : dec_tgt[b * T_ + (t - 1)];
  for (int c = tid; c < RINP_; c += 256) {
    float v = 0.f;
    if (c < 256) v = wordemb[tok * HID_ + c];
    else if (c < 552) {
      int cc = c - 256;
      float acc = 0.f;
      for (int l = 0; l < L_; ++l)
        acc += w[l] * (float)enc[((size_t)b * L_ + l) * CEP_ + cc];
      v = acc;
    }
    rin[(size_t)b * RINP_ + c] = (h16)v;
  }
}

__global__ void gate_k(const float* __restrict__ gi, const float* __restrict__ gh,
                       const float* __restrict__ bih, const float* __restrict__ bhh,
                       float* __restrict__ hidden, h16* __restrict__ hh16) {
  int b = blockIdx.x, h = threadIdx.x;
  size_t o = (size_t)b * G3_;
  float ir = gi[o + h]        + bih[h],        hr = gh[o + h]        + bhh[h];
  float iz = gi[o + 256 + h]  + bih[256 + h],  hz = gh[o + 256 + h]  + bhh[256 + h];
  float in = gi[o + 512 + h]  + bih[512 + h],  hn = gh[o + 512 + h]  + bhh[512 + h];
  float r = 1.f / (1.f + expf(-(ir + hr)));
  float z = 1.f / (1.f + expf(-(iz + hz)));
  float n = tanhf(in + r * hn);
  size_t hi = (size_t)b * HID_ + h;
  float hnew = (1.f - z) * n + z * hidden[hi];
  hidden[hi] = hnew;
  hh16[hi] = (h16)hnew;
}

__global__ void nll_k(const float* __restrict__ logits, const float* __restrict__ out_b,
                      const int* __restrict__ word_tgt, int t, float* __restrict__ nllacc) {
  int b = blockIdx.x, lane = threadIdx.x;
  const float* lg = logits + (size_t)b * NCP_;
  float mx = -1e30f;
  for (int c = lane; c < NC_; c += 32) mx = fmaxf(mx, lg[c] + out_b[c]);
#pragma unroll
  for (int off = 16; off; off >>= 1) mx = fmaxf(mx, __shfl_xor(mx, off, 32));
  float s = 0.f;
  for (int c = lane; c < NC_; c += 32) s += expf(lg[c] + out_b[c] - mx);
#pragma unroll
  for (int off = 16; off; off >>= 1) s += __shfl_xor(s, off, 32);
  if (lane == 0) {
    int tgt = word_tgt[b * T_ + t];
    float lse = mx + logf(s);
    float nll = 0.f;
    if (tgt >= 0) nll = lse - (lg[tgt] + out_b[tgt]);
    nllacc[b] += nll;
  }
}

__global__ void loss_k(const float* __restrict__ nllacc, float* __restrict__ out) {
  __shared__ float red[128];
  int tid = threadIdx.x;
  red[tid] = nllacc[tid]; __syncthreads();
  for (int off = 64; off; off >>= 1) { if (tid < off) red[tid] += red[tid + off]; __syncthreads(); }
  if (tid == 0) out[0] = 0.2f * (red[0] / (float)B_);
}

// ---------------------------------------------------------------------------
static inline size_t alignup(size_t x) { return (x + 255) & ~(size_t)255; }

extern "C" void kernel_launch(void* const* d_in, const int* in_sizes, int n_in,
                              void* d_out, int out_size, void* d_ws, size_t ws_size,
                              hipStream_t stream) {
  (void)in_sizes; (void)n_in; (void)out_size; (void)ws_size;
  const float* x        = (const float*)d_in[0];
  const int*   dec_tgt  = (const int*)  d_in[1];
  const int*   word_tgt = (const int*)  d_in[2];
  const float* conv_w   = (const float*)d_in[3];
  const float* conv_b   = (const float*)d_in[4];
  const float* emb_dec  = (const float*)d_in[5];
  const float* word_w   = (const float*)d_in[6];
  const float* word_b   = (const float*)d_in[7];
  const float* attn_w   = (const float*)d_in[8];
  const float* attn_b   = (const float*)d_in[9];
  const float* attn_v   = (const float*)d_in[10];
  const float* gru_wih  = (const float*)d_in[11];
  const float* gru_bih  = (const float*)d_in[12];
  const float* gru_whh  = (const float*)d_in[13];
  const float* gru_bhh  = (const float*)d_in[14];
  const float* out_w    = (const float*)d_in[15];
  const float* out_b    = (const float*)d_in[16];
  const float* x_emb    = (const float*)d_in[17];
  const float* y_emb    = (const float*)d_in[18];
  float* loss_out = (float*)d_out;

  // ---- workspace carve (~80 MB) ----
  char* p = (char*)d_ws;
  auto carve = [&](size_t bytes) { void* q = (void*)p; p += alignup(bytes); return q; };
  h16*   xh      = (h16*)  carve((size_t)B_ * 256 * 8 * 32 * 2);        // 16.8MB
  h16*   wparT   = (h16*)  carve((size_t)4 * 256 * KCONV_ * 2);        // 4.7MB
  h16*   attnTe  = (h16*)  carve((size_t)HID_ * CEP_ * 2);
  h16*   attnTh  = (h16*)  carve((size_t)HID_ * HID_ * 2);
  h16*   gwihT   = (h16*)  carve((size_t)G3_ * RINP_ * 2);
  h16*   gwhhT   = (h16*)  carve((size_t)G3_ * HID_ * 2);
  h16*   outT    = (h16*)  carve((size_t)NCP_ * HID_ * 2);
  float* wemb    = (float*)carve((size_t)NC_ * HID_ * 4);
  h16*   enc     = (h16*)  carve((size_t)B_ * L_ * CEP_ * 2);          // 21MB
  float* encA    = (float*)carve((size_t)B_ * L_ * HID_ * 4);          // 33.5MB
  float* hidden  = (float*)carve((size_t)B_ * HID_ * 4);
  h16*   hh16    = (h16*)  carve((size_t)B_ * HID_ * 2);
  float* hterm   = (float*)carve((size_t)B_ * HID_ * 4);
  float* scores  = (float*)carve((size_t)B_ * L_ * 4);
  h16*   rin     = (h16*)  carve((size_t)B_ * RINP_ * 2);
  float* gi      = (float*)carve((size_t)B_ * G3_ * 4);
  float* gh      = (float*)carve((size_t)B_ * G3_ * 4);
  float* logits  = (float*)carve((size_t)B_ * NCP_ * 4);
  float* nllacc  = (float*)carve((size_t)B_ * 4);

  // ---- prep ----
  int nx = B_ * 256 * 8 * 32;
  cast_x_k<<<(nx + 255) / 256, 256, 0, stream>>>(x, xh, nx);
  wpar_k<<<(4 * 256 * 256 * 9) / 256, 256, 0, stream>>>(conv_w, wparT);
  pack_f16_k<<<(HID_ * CEP_) / 256, 256, 0, stream>>>(attn_w, attnTe, HID_, HID_, CEP_, CE_, 552, 256);
  pack_f16_k<<<(HID_ * HID_) / 256, 256, 0, stream>>>(attn_w, attnTh, HID_, HID_, HID_, HID_, 552, 0);
  pack_f16_k<<<(G3_ * RINP_) / 256, 256, 0, stream>>>(gru_wih, gwihT, G3_, G3_, RINP_, 552, 552, 0);
  pack_f16_k<<<(G3_ * HID_) / 256, 256, 0, stream>>>(gru_whh, gwhhT, G3_, G3_, HID_, HID_, HID_, 0);
  pack_f16_k<<<(NCP_ * HID_) / 256, 256, 0, stream>>>(out_w, outT, NCP_, NC_, HID_, HID_, HID_, 0);
  wordemb_k<<<NC_, HID_, 0, stream>>>(emb_dec, word_w, word_b, wemb);
  posfill_k<<<(B_ * L_ * 64) / 256, 256, 0, stream>>>(x_emb, y_emb, enc);
  hipMemsetAsync(hidden, 0, (size_t)B_ * HID_ * 4, stream);
  hipMemsetAsync(hh16,   0, (size_t)B_ * HID_ * 2, stream);
  hipMemsetAsync(nllacc, 0, (size_t)B_ * 4, stream);

  // ---- encoder: folded conv GEMM (+pool+relu) then attention projection ----
  conv_gemm_k<<<dim3(2048, 2), 256, 0, stream>>>(xh, wparT, conv_b, enc);
  gemm16_k<<<dim3((B_ * L_) / 16, HID_ / 16), 32, 0, stream>>>(
      enc, attnTe, encA, attn_b, CEP_, HID_, CEP_ / 32);

  // ---- decoder scan ----
  for (int t = 0; t < T_; ++t) {
    gemm16_k<<<dim3(B_ / 16, HID_ / 16), 32, 0, stream>>>(
        hh16, attnTh, hterm, nullptr, HID_, HID_, HID_ / 32);
    scores_k<<<B_, 256, 0, stream>>>(encA, hterm, attn_v, scores);
    softctx_k<<<B_, 256, 0, stream>>>(scores, enc, wemb, dec_tgt, t, rin);
    gemm16_k<<<dim3(B_ / 16, G3_ / 16), 32, 0, stream>>>(
        rin, gwihT, gi, nullptr, RINP_, G3_, RINP_ / 32);
    gemm16_k<<<dim3(B_ / 16, G3_ / 16), 32, 0, stream>>>(
        hh16, gwhhT, gh, nullptr, HID_, G3_, HID_ / 32);
    gate_k<<<B_, HID_, 0, stream>>>(gi, gh, gru_bih, gru_bhh, hidden, hh16);
    gemm16_k<<<dim3(B_ / 16, NCP_ / 16), 32, 0, stream>>>(
        hh16, outT, logits, nullptr, HID_, NCP_, HID_ / 32);
    nll_k<<<B_, 32, 0, stream>>>(logits, out_b, word_tgt, t, nllacc);
  }
  loss_k<<<1, 128, 0, stream>>>(nllacc, loss_out);
}
